// DecentNeRFField_79370995630435
// MI455X (gfx1250) — matrix operand order repaired and checked by reading, not compiled, not run on hardware
//
#include <hip/hip_runtime.h>

typedef __attribute__((ext_vector_type(16))) _Float16 v16h;
typedef __attribute__((ext_vector_type(8)))  float    v8f;

#define NT 512   // 16 waves; 16 points per wave -> 256 points/block

// ---------------------------------------------------------------------------
// d_ws layout: pre-swizzled f16 A-fragment blobs, then padded f32 biases.
// ---------------------------------------------------------------------------
enum : int {
  OFF_PW0   = 0,        // KT=2, MT=8 ->  8192 halves
  OFF_PW1   = 8192,     // KT=4, MT=8 -> 16384
  OFF_PW2   = 24576,
  OFF_PW3   = 40960,
  OFF_PW4   = 57344,    // KT=6, MT=8 -> 24576
  OFF_PW5   = 81920,
  OFF_PW6   = 98304,
  OFF_PW7   = 114688,
  OFF_FEAT  = 131072,   // KT=4, MT=8 -> 16384
  OFF_RGB1  = 147456,   // KT=5, MT=4 -> 10240
  OFF_SMALL = 157696,   // resident heads, 13312 halves total
  OFF_SIGMA = OFF_SMALL + 0,
  OFF_RGB2  = OFF_SMALL + 2048,
  OFF_M1    = OFF_SMALL + 3072,
  OFF_M2    = OFF_SMALL + 7168,
  OFF_RGBH  = OFF_SMALL + 11264,
  OFF_SIGH  = OFF_SMALL + 12288,
  WS_HALFS  = 171008,
};
enum : int {           // float offsets in the bias blob
  BOFF_PB   = 0,       // 8 x 128
  BOFF_FEAT = 1024,
  BOFF_RGB1 = 1152,    // 64
  BOFF_SB   = 1216,    // 192-entry small-bias block
  WS_FLOATS = 1408,
};

// ---------------------------------------------------------------------------
// Prep kernels: swizzle W (fan_in x fan_out f32, row-major) into f16
// A-fragments of W^T. Fragment (mt,kt): outs [16mt,16mt+16) x ins [32kt,32kt+32).
// A-layout used here (contiguous k, B matches): lane l<16: M=l, halves0-15->K=h;
// lane 16+l: K += 16. Per fragment: 32 lanes x 16 halves contiguous (32B/lane).
// ---------------------------------------------------------------------------
__global__ void swizzle_w_kernel(const float* __restrict__ W, int fan_in,
                                 int fan_out, int KT, int MT,
                                 _Float16* __restrict__ dst) {
  int total = MT * KT * 512;
  for (int i = blockIdx.x * blockDim.x + threadIdx.x; i < total;
       i += gridDim.x * blockDim.x) {
    int h    = i & 15;
    int lane = (i >> 4) & 31;
    int frag = i >> 9;
    int kt   = frag % KT;
    int mt   = frag / KT;
    int m = (lane & 15) + mt * 16;
    int k = h + ((lane & 16) ? 16 : 0) + kt * 32;
    float w = (k < fan_in && m < fan_out) ? W[k * fan_out + m] : 0.0f;
    dst[i] = (_Float16)w;
  }
}

__global__ void pad_bias_kernel(const float* __restrict__ b, int n, int npad,
                                float* __restrict__ dst) {
  int i = threadIdx.x;
  if (i < npad) dst[i] = (i < n) ? b[i] : 0.0f;
}

// ---------------------------------------------------------------------------
// Main-kernel helpers
// ---------------------------------------------------------------------------
__device__ __forceinline__ v8f wmma_f16(v16h a, v16h b, v8f c) {
  return __builtin_amdgcn_wmma_f32_16x16x32_f16(false, a, false, b, (short)0, c,
                                                false, false);
}

// Async DMA: global (pre-swizzled f16) -> LDS, 16B per lane per op (ASYNCcnt).
__device__ __forceinline__ void async_copy_lds(const _Float16* __restrict__ src,
                                               const _Float16* dst_lds,
                                               int halfs, int tid) {
  const int bytes = halfs * 2;
  uint32_t lbase = (uint32_t)(uintptr_t)dst_lds;  // LDS aperture low 32 bits
  for (int off = tid * 16; off < bytes; off += NT * 16) {
    uint32_t laddr = lbase + (uint32_t)off;
    asm volatile("global_load_async_to_lds_b128 %0, %1, %2"
                 :: "v"(laddr), "v"(off), "s"(src)
                 : "memory");
  }
}

__device__ __forceinline__ void wait_async0() {
  asm volatile("s_wait_asynccnt 0x0" ::: "memory");
}

// D layout (16x16 f32): lane n (<16): VGPR j = D[m=j][n]; lane n+16: m=8+j.
__device__ __forceinline__ void bias_act(v8f& D, const float* bias, int mtile,
                                         int lane, bool relu) {
  int off = mtile * 16 + ((lane & 16) ? 8 : 0);
#pragma unroll
  for (int j = 0; j < 8; ++j) {
    float v = D[j] + bias[off + j];
    D[j] = relu ? fmaxf(v, 0.0f) : v;
  }
}

// D0 (features f0..f0+15) + D1 (f0+16..f0+31) -> next-layer B fragment.
__device__ __forceinline__ v16h combine_frag(v8f D0, v8f D1, int lane) {
  union H4 { int i[4]; _Float16 f[8]; };
  H4 pe, pf;
#pragma unroll
  for (int j = 0; j < 8; ++j) {
    pe.f[j] = (_Float16)D0[j];
    pf.f[j] = (_Float16)D1[j];
  }
  const bool hi = (lane & 16) != 0;
  union R { v16h h; int i[8]; } r;
#pragma unroll
  for (int j = 0; j < 4; ++j) {
    int send = hi ? pe.i[j] : pf.i[j];
    int recv = __shfl_xor(send, 16, 32);
    r.i[j]     = hi ? recv    : pe.i[j];
    r.i[4 + j] = hi ? pf.i[j] : recv;
  }
  return r.h;
}

template <int KT, bool RELU>
__device__ __forceinline__ v16h pair_tile(const _Float16* wb, int mt,
                                          const v16h* Bin, const float* bias,
                                          int lane) {
  v8f D0 = {}, D1 = {};
#pragma unroll
  for (int kt = 0; kt < KT; ++kt) {
    v16h a0 = *(const v16h*)(wb + (((mt    ) * KT + kt) * 32 + lane) * 16);
    v16h a1 = *(const v16h*)(wb + (((mt + 1) * KT + kt) * 32 + lane) * 16);
    D0 = wmma_f16(a0, Bin[kt], D0);
    D1 = wmma_f16(a1, Bin[kt], D1);
  }
  bias_act(D0, bias, mt, lane, RELU);
  bias_act(D1, bias, mt + 1, lane, RELU);
  return combine_frag(D0, D1, lane);
}

template <int KT>
__device__ __forceinline__ v8f head_tile(const _Float16* wb, const v16h* Bin,
                                         int lane) {
  v8f D = {};
#pragma unroll
  for (int kt = 0; kt < KT; ++kt) {
    v16h a = *(const v16h*)(wb + (kt * 32 + lane) * 16);
    D = wmma_f16(a, Bin[kt], D);
  }
  return D;
}

// Positional encoding generated directly in B-fragment layout.
__device__ __forceinline__ float enc_feat(float cx, float cy, float cz, int k,
                                          int nfeat) {
  if (k >= nfeat) return 0.0f;
  if (k < 3) return k == 0 ? cx : (k == 1 ? cy : cz);
  int j = k - 3;
  int fr = j / 6;
  int rem = j % 6;
  int c = rem % 3;
  float v = (c == 0 ? cx : (c == 1 ? cy : cz)) * (float)(1 << fr);
  return (rem < 3) ? __sinf(v) : __cosf(v);
}

__device__ __forceinline__ v16h enc_frag(float cx, float cy, float cz, int ktile,
                                         int lane, int nfeat) {
  v16h r;
  int kbase = ktile * 32 + ((lane & 16) ? 16 : 0);
#pragma unroll
  for (int h = 0; h < 16; ++h)
    r[h] = (_Float16)enc_feat(cx, cy, cz, kbase + h, nfeat);
  return r;
}

struct KParams {
  const float* x; const float* d; const int* img_idx;
  const float* app_emb; const float* trans_emb;
  const _Float16* w;   // swizzled fragment blob (d_ws)
  const float* b;      // padded bias blob (d_ws tail)
  float* out; int N;
};

// Dense layer -> 128 outs. Single-stage: async DMA whole layer, then 4 pairs.
template <int KT, bool RELU>
__device__ __forceinline__ void layer128(const _Float16* wsrc, const float* bsrc,
                                         v16h* B, _Float16* s_w, float* s_bias,
                                         int tid, int lane) {
  __syncthreads();                       // previous users of s_w are done
  async_copy_lds(wsrc, s_w, KT * 8 * 512, tid);
  if (tid < 128) s_bias[tid] = bsrc[tid];
  wait_async0();
  __syncthreads();
  v16h n0 = pair_tile<KT, RELU>(s_w, 0, B, s_bias, lane);
  v16h n1 = pair_tile<KT, RELU>(s_w, 2, B, s_bias, lane);
  v16h n2 = pair_tile<KT, RELU>(s_w, 4, B, s_bias, lane);
  v16h n3 = pair_tile<KT, RELU>(s_w, 6, B, s_bias, lane);
  B[0] = n0; B[1] = n1; B[2] = n2; B[3] = n3;
}

__global__ __launch_bounds__(NT) void decent_nerf_kernel(KParams P) {
  __shared__ alignas(32) _Float16 s_w[24576];      // 48KB per-layer staging
  __shared__ alignas(32) _Float16 s_small[13312];  // 26KB resident small heads
  __shared__ float s_bias[128];
  __shared__ float s_sbias[192];

  const int tid  = threadIdx.x;
  const int lane = tid & 31;
  const int wave = tid >> 5;
  const int N    = P.N;

  // resident heads: async DMA once; first layer's wait+barrier covers it
  async_copy_lds(P.w + OFF_SMALL, s_small, 13312, tid);
  if (tid < 192) s_sbias[tid] = P.b[BOFF_SB + tid];

  // per-wave tile of 16 points; lanes n and n+16 both serve point n
  int point = blockIdx.x * (NT / 2) + wave * 16 + (lane & 15);
  int pc = point < N ? point : N - 1;

  float cx = P.x[3 * pc + 0], cy = P.x[3 * pc + 1], cz = P.x[3 * pc + 2];
  float dx = P.d[3 * pc + 0], dy = P.d[3 * pc + 1], dz = P.d[3 * pc + 2];

  v16h xe0 = enc_frag(cx, cy, cz, 0, lane, 63);
  v16h xe1 = enc_frag(cx, cy, cz, 1, lane, 63);
  v16h de  = enc_frag(dx, dy, dz, 0, lane, 27);

  v16h B[6];
  B[0] = xe0; B[1] = xe1;

  // pts MLP: 8 layers, skip-concat of x_enc at layer 4
  layer128<2, true>(P.w + OFF_PW0, P.b + 0 * 128, B, s_w, s_bias, tid, lane);
  layer128<4, true>(P.w + OFF_PW1, P.b + 1 * 128, B, s_w, s_bias, tid, lane);
  layer128<4, true>(P.w + OFF_PW2, P.b + 2 * 128, B, s_w, s_bias, tid, lane);
  layer128<4, true>(P.w + OFF_PW3, P.b + 3 * 128, B, s_w, s_bias, tid, lane);
  B[4] = xe0; B[5] = xe1;                        // concat(h, x_enc) -> 192
  layer128<6, true>(P.w + OFF_PW4, P.b + 4 * 128, B, s_w, s_bias, tid, lane);
  layer128<4, true>(P.w + OFF_PW5, P.b + 5 * 128, B, s_w, s_bias, tid, lane);
  layer128<4, true>(P.w + OFF_PW6, P.b + 6 * 128, B, s_w, s_bias, tid, lane);
  layer128<4, true>(P.w + OFF_PW7, P.b + 7 * 128, B, s_w, s_bias, tid, lane);

  // sigma head: 128 -> 1 (padded to 16), relu
  v8f Ds = head_tile<4>(s_small + (OFF_SIGMA - OFF_SMALL), B, lane);
  float sigma_g = fmaxf(Ds[0] + s_sbias[0], 0.0f);  // valid at lanes 0-15

  // feat: 128 -> 128, bias only
  layer128<4, false>(P.w + OFF_FEAT, P.b + BOFF_FEAT, B, s_w, s_bias, tid, lane);
  B[4] = de;                                     // concat(feat, d_enc) -> 160

  // rgb1: 155(->160) -> 64, relu (KT=5, MT=4, one stage)
  __syncthreads();
  async_copy_lds(P.w + OFF_RGB1, s_w, 5 * 4 * 512, tid);
  if (tid < 64) s_bias[tid] = P.b[BOFF_RGB1 + tid];
  wait_async0();
  __syncthreads();
  v16h C[2];
  C[0] = pair_tile<5, true>(s_w, 0, B, s_bias, lane);
  C[1] = pair_tile<5, true>(s_w, 2, B, s_bias, lane);

  // rgb2: 64 -> 3, sigmoid
  v8f Dr = head_tile<2>(s_small + (OFF_RGB2 - OFF_SMALL), C, lane);
  float rg0 = 1.0f / (1.0f + __expf(-(Dr[0] + s_sbias[16 + 0])));
  float rg1 = 1.0f / (1.0f + __expf(-(Dr[1] + s_sbias[16 + 1])));
  float rg2 = 1.0f / (1.0f + __expf(-(Dr[2] + s_sbias[16 + 2])));

  // personal head: concat(app[idx](32), trans[idx](16)) -> 48 (pad 64)
  int idx = P.img_idx[pc];
  const float* ae = P.app_emb   + idx * 32;
  const float* te = P.trans_emb + idx * 16;
  v16h Pp[2];
#pragma unroll
  for (int kt = 0; kt < 2; ++kt) {
    v16h r;
    int kbase = kt * 32 + ((lane & 16) ? 16 : 0);
#pragma unroll
    for (int h = 0; h < 16; ++h) {
      int k = kbase + h;
      float v = (k < 32) ? ae[k] : ((k < 48) ? te[k - 32] : 0.0f);
      r[h] = (_Float16)v;
    }
    Pp[kt] = r;
  }
  v16h Q[2];
  Q[0] = pair_tile<2, true>(s_small + (OFF_M1 - OFF_SMALL), 0, Pp, s_sbias + 32, lane);
  Q[1] = pair_tile<2, true>(s_small + (OFF_M1 - OFF_SMALL), 2, Pp, s_sbias + 32, lane);
  v16h Rr[2];
  Rr[0] = pair_tile<2, true>(s_small + (OFF_M2 - OFF_SMALL), 0, Q, s_sbias + 96, lane);
  Rr[1] = pair_tile<2, true>(s_small + (OFF_M2 - OFF_SMALL), 2, Q, s_sbias + 96, lane);

  v8f Dh  = head_tile<2>(s_small + (OFF_RGBH - OFF_SMALL), Rr, lane);
  v8f Dsg = head_tile<2>(s_small + (OFF_SIGH - OFF_SMALL), Rr, lane);
  float rp0 = 1.0f / (1.0f + __expf(-(Dh[0] + s_sbias[160 + 0])));
  float rp1 = 1.0f / (1.0f + __expf(-(Dh[1] + s_sbias[160 + 1])));
  float rp2 = 1.0f / (1.0f + __expf(-(Dh[2] + s_sbias[160 + 2])));
  float sigma_p = fmaxf(Dsg[0] + s_sbias[176], 0.0f);

  if (point < N && !(lane & 16)) {
    float Tg = __expf(-sigma_g);
    float om = 1.0f - Tg;
    P.out[3 * point + 0] = Tg * rp0 + om * rg0;
    P.out[3 * point + 1] = Tg * rp1 + om * rg1;
    P.out[3 * point + 2] = Tg * rp2 + om * rg2;
    P.out[3 * N + point] = sigma_g + sigma_p;
  }
}

// ---------------------------------------------------------------------------
extern "C" void kernel_launch(void* const* d_in, const int* in_sizes, int n_in,
                              void* d_out, int out_size, void* d_ws,
                              size_t ws_size, hipStream_t stream) {
  (void)n_in; (void)out_size; (void)ws_size;
  const float* x   = (const float*)d_in[0];
  const float* d   = (const float*)d_in[1];
  const int* idx   = (const int*)  d_in[2];
  const float* pw[8]; const float* pb[8];
  for (int L = 0; L < 8; ++L) pw[L] = (const float*)d_in[3 + L];
  for (int L = 0; L < 8; ++L) pb[L] = (const float*)d_in[11 + L];
  const float* sigma_w = (const float*)d_in[19]; const float* sigma_b = (const float*)d_in[20];
  const float* feat_w  = (const float*)d_in[21]; const float* feat_b  = (const float*)d_in[22];
  const float* rgb1_w  = (const float*)d_in[23]; const float* rgb1_b  = (const float*)d_in[24];
  const float* rgb2_w  = (const float*)d_in[25]; const float* rgb2_b  = (const float*)d_in[26];
  const float* app     = (const float*)d_in[27]; const float* trans   = (const float*)d_in[28];
  const float* m1w = (const float*)d_in[29]; const float* m1b = (const float*)d_in[30];
  const float* m2w = (const float*)d_in[31]; const float* m2b = (const float*)d_in[32];
  const float* rhw = (const float*)d_in[33]; const float* rhb = (const float*)d_in[34];
  const float* shw = (const float*)d_in[35]; const float* shb = (const float*)d_in[36];

  _Float16* wsw = (_Float16*)d_ws;
  float*    wsb = (float*)((char*)d_ws + (size_t)WS_HALFS * 2);

  // ---- prep: swizzle weights to f16 fragments, pad biases (every launch) ----
  struct { const float* W; int fi, fo, KT, MT, off; } sw[] = {
    {pw[0],    63, 128, 2, 8, OFF_PW0}, {pw[1],  128, 128, 4, 8, OFF_PW1},
    {pw[2],   128, 128, 4, 8, OFF_PW2}, {pw[3],  128, 128, 4, 8, OFF_PW3},
    {pw[4],   191, 128, 6, 8, OFF_PW4}, {pw[5],  128, 128, 4, 8, OFF_PW5},
    {pw[6],   128, 128, 4, 8, OFF_PW6}, {pw[7],  128, 128, 4, 8, OFF_PW7},
    {feat_w,  128, 128, 4, 8, OFF_FEAT},{rgb1_w, 155,  64, 5, 4, OFF_RGB1},
    {sigma_w, 128,   1, 4, 1, OFF_SIGMA},{rgb2_w,  64,   3, 2, 1, OFF_RGB2},
    {m1w,      48,  64, 2, 4, OFF_M1},  {m2w,     64,  64, 2, 4, OFF_M2},
    {rhw,      64,   3, 2, 1, OFF_RGBH},{shw,     64,   1, 2, 1, OFF_SIGH},
  };
  for (auto& s : sw) {
    int total = s.MT * s.KT * 512;
    hipLaunchKernelGGL(swizzle_w_kernel, dim3((total + 255) / 256), dim3(256), 0,
                       stream, s.W, s.fi, s.fo, s.KT, s.MT, wsw + s.off);
  }
  struct { const float* b; int n, npad, boff; } bj[] = {
    {pb[0], 128, 128, 0 * 128}, {pb[1], 128, 128, 1 * 128},
    {pb[2], 128, 128, 2 * 128}, {pb[3], 128, 128, 3 * 128},
    {pb[4], 128, 128, 4 * 128}, {pb[5], 128, 128, 5 * 128},
    {pb[6], 128, 128, 6 * 128}, {pb[7], 128, 128, 7 * 128},
    {feat_b, 128, 128, BOFF_FEAT}, {rgb1_b, 64, 64, BOFF_RGB1},
    {sigma_b, 1, 16, BOFF_SB + 0},  {rgb2_b, 3, 16, BOFF_SB + 16},
    {m1b, 64, 64, BOFF_SB + 32},    {m2b, 64, 64, BOFF_SB + 96},
    {rhb, 3, 16, BOFF_SB + 160},    {shb, 1, 16, BOFF_SB + 176},
  };
  for (auto& s : bj)
    hipLaunchKernelGGL(pad_bias_kernel, dim3(1), dim3(256), 0, stream,
                       s.b, s.n, s.npad, wsb + s.boff);

  // ---- main fused kernel ----
  KParams P;
  P.x = x; P.d = d; P.img_idx = idx;
  P.app_emb = app; P.trans_emb = trans;
  P.w = wsw; P.b = wsb;
  P.out = (float*)d_out;
  int N = in_sizes[0] / 3;
  P.N = N;
  int blocks = (N + (NT / 2) - 1) / (NT / 2);
  hipLaunchKernelGGL(decent_nerf_kernel, dim3(blocks), dim3(NT), 0, stream, P);
}
